// RhnLin_21225728377123
// MI455X (gfx1250) — matrix operand order repaired and checked by reading, not compiled, not run on hardware
//
#include <hip/hip_runtime.h>
#include <math.h>

// ---------------------------------------------------------------------------
// RHN (Recurrent Highway Network) forward for MI455X / gfx1250 (wave32, WMMA)
//
//  B=16, T=1024, I=H=1024, L=5.
//  Strategy:
//   * All GEMMs in bf16 via v_wmma_f32_16x16x32_bf16 (M=16 == batch).
//   * Recurrent weights converted once to bf16 (30 MB -> resident in 192MB L2).
//   * Input projections precomputed in a fully parallel WMMA kernel.
//   * Sequential scan as a persistent kernel: 64 blocks (one per 16-col tile),
//     4 waves/block splitting K=1024, LDS reduction, global atomic barrier
//     between the 5120 micro-layers.
// ---------------------------------------------------------------------------

#define Lc 5
#define Hc 1024
#define Ic 1024
#define Bc 16
#define Tc 1024

#define NBLK 64           // persistent blocks in the recurrence (must co-reside)
#define RECUR_THREADS 128 // 4 waves of 32

typedef __attribute__((ext_vector_type(16))) __bf16 v16bf;
typedef __attribute__((ext_vector_type(8)))  float  v8f;

static __device__ inline v8f vzero8() {
  v8f z = {0.f, 0.f, 0.f, 0.f, 0.f, 0.f, 0.f, 0.f};
  return z;
}

static __device__ inline v8f wmma_bf16(v16bf a, v16bf b, v8f c) {
  // D = A(16x32) * B(32x16) + C(16x16 f32)
  return __builtin_amdgcn_wmma_f32_16x16x32_bf16(
      /*neg_a=*/false, a, /*neg_b=*/false, b,
      /*c_mod=*/(short)0, c, /*reuse_a=*/false, /*reuse_b=*/false);
}

// B fragment (K=32 x N=16) from bf16 row-major W[n][k], leading dim LD.
// ISA layout: lane l holds column n = l&15; with ko = (l>>4)*8,
// halves 0..7 = K kb+ko..kb+ko+7 ; halves 8..15 = K kb+16+ko..kb+16+ko+7.
static __device__ inline v16bf load_b_frag(const __bf16* __restrict__ W,
                                           int nbase, int kb, int LD, int lane) {
  typedef __attribute__((ext_vector_type(8))) __bf16 v8bf;
  const int n  = nbase + (lane & 15);
  const int kg = (lane >> 4) << 3;
  const __bf16* p = W + (size_t)n * LD + kb + kg;
  v8bf lo = *(const v8bf*)p;
  v8bf hi = *(const v8bf*)(p + 16);
  return __builtin_shufflevector(lo, hi, 0, 1, 2, 3, 4, 5, 6, 7, 8, 9, 10, 11, 12, 13, 14, 15);
}

// A fragment (M=16 x K=32) from f32 row-major S[m][k] (ld = LD), cvt -> bf16.
// ISA layout: lane l holds row m = l&15; with ko = (l>>4)*8,
// halves 0..7 = K kb+ko..kb+ko+7 ; halves 8..15 = K kb+16+ko..kb+16+ko+7.
static __device__ inline v16bf load_a_frag_f32(const float* __restrict__ S,
                                               int kb, int LD, int lane) {
  const int m  = lane & 15;
  const int ko = (lane >> 4) << 3;
  const float* p = S + (size_t)m * LD + kb + ko;
  v16bf a;
#pragma unroll
  for (int i = 0; i < 8; ++i) {
    a[i]     = (__bf16)p[i];
    a[i + 8] = (__bf16)p[i + 16];
  }
  return a;
}

// ---------------------------------------------------------------------------
// Pack kernels (one-time per launch, fully parallel, bandwidth-trivial)
// ---------------------------------------------------------------------------

// R_g[l][n][k] f32 -> Rpack[g][l][n][k] bf16 (layout already B-frag friendly:
// the recurrence needs B[k][n] = R[n][k], i.e. row-major R itself).
__global__ void pack_R(const float* __restrict__ Rh, const float* __restrict__ Rt,
                       const float* __restrict__ Rc, __bf16* __restrict__ Rp) {
  const size_t NT = (size_t)Lc * Hc * Hc;
  for (size_t i = (size_t)blockIdx.x * blockDim.x + threadIdx.x; i < NT;
       i += (size_t)gridDim.x * blockDim.x) {
    Rp[i]          = (__bf16)Rh[i];
    Rp[NT + i]     = (__bf16)Rt[i];
    Rp[2 * NT + i] = (__bf16)Rc[i];
  }
}

// w_g[k][n] f32 -> wT[g][n][k] bf16 (so the proj kernel's B-frag loads are
// contiguous per lane, same as R).
__global__ void pack_wT(const float* __restrict__ wh, const float* __restrict__ wt,
                        const float* __restrict__ wc, __bf16* __restrict__ wT) {
  const size_t NT = (size_t)Hc * Ic;
  for (size_t i = (size_t)blockIdx.x * blockDim.x + threadIdx.x; i < NT;
       i += (size_t)gridDim.x * blockDim.x) {
    const size_t n = i / Ic, k = i % Ic;
    const size_t src = k * Hc + n;
    wT[i]          = (__bf16)wh[src];
    wT[NT + i]     = (__bf16)wt[src];
    wT[2 * NT + i] = (__bf16)wc[src];
  }
}

// ---------------------------------------------------------------------------
// Input projections: proj[g][t][b][n] = sum_k x[b][t][k] * w_g[k][n]   (bf16)
// One wave per (t, g, 64-col tile); A fragment (x rows) reused across 4 tiles.
// ---------------------------------------------------------------------------
__global__ __launch_bounds__(32) void rhn_proj(const float* __restrict__ x,
                                               const __bf16* __restrict__ wT,
                                               __bf16* __restrict__ proj) {
  const int lane  = threadIdx.x & 31;
  const int t     = blockIdx.x;
  const int g     = blockIdx.y;
  const int nbase = blockIdx.z * 64;
  const __bf16* W = wT + (size_t)g * Hc * Ic;

  v8f acc[4] = {vzero8(), vzero8(), vzero8(), vzero8()};
  const int m  = lane & 15;
  const int ko = (lane >> 4) << 3;
  const float* xrow = x + ((size_t)m * Tc + t) * Ic;  // x[b=m][t][*]

  for (int kb = 0; kb < Ic; kb += 32) {
    const float* p = xrow + kb + ko;
    v16bf a;
#pragma unroll
    for (int i = 0; i < 8; ++i) {
      a[i]     = (__bf16)p[i];
      a[i + 8] = (__bf16)p[i + 16];
    }
#pragma unroll
    for (int nt = 0; nt < 4; ++nt) {
      v16bf b = load_b_frag(W, nbase + nt * 16, kb, Ic, lane);
      acc[nt] = wmma_bf16(a, b, acc[nt]);
    }
  }

  const int n0 = lane & 15;
  const int hi = lane >> 4;
#pragma unroll
  for (int nt = 0; nt < 4; ++nt)
#pragma unroll
    for (int v = 0; v < 8; ++v) {
      const int mm = v + hi * 8;  // C/D layout: m = v + (lane>=16)*8
      proj[(((size_t)g * Tc + t) * Bc + mm) * Hc + (nbase + nt * 16 + n0)] =
          (__bf16)acc[nt][v];
    }
}

// ---------------------------------------------------------------------------
// Persistent recurrence kernel. 64 blocks x 4 waves. Block b owns output
// columns [16b, 16b+16). Each wave handles a K-slice of 256 for all 3 gates;
// partial 16x16 f32 tiles reduced via LDS; wave 0 applies gates and writes
// the new state. Global atomic barrier separates the 5120 micro-layers.
// ---------------------------------------------------------------------------
__global__ __launch_bounds__(RECUR_THREADS) void rhn_recur(
    const __bf16* __restrict__ Rpack,  // [3][L][H][H] bf16
    const __bf16* __restrict__ proj,   // [3][T][B][H] bf16
    const float* __restrict__ bh, const float* __restrict__ bt,
    const float* __restrict__ bc,      // [L][H] f32
    float* __restrict__ sb0, float* __restrict__ sb1,  // [B][H] f32 x2
    float* __restrict__ out,           // [B][T][H]
    float* __restrict__ sfin,          // [B][H]
    unsigned* __restrict__ counter) {
  const int lane  = threadIdx.x & 31;
  const int wave  = threadIdx.x >> 5;
  const int nbase = blockIdx.x * 16;

  __shared__ float red[4][3][256];  // [wave][gate][v*32+lane]

  float* scur = sb0;
  float* snxt = sb1;
  unsigned gen = 0;

  for (int t = 0; t < Tc; ++t) {
    for (int l = 0; l < Lc; ++l) {
      const int k0 = wave * (Hc / 4);
      v8f acc0 = vzero8(), acc1 = vzero8(), acc2 = vzero8();
      const __bf16* W0 = Rpack + (size_t)(0 * Lc + l) * Hc * Hc;
      const __bf16* W1 = Rpack + (size_t)(1 * Lc + l) * Hc * Hc;
      const __bf16* W2 = Rpack + (size_t)(2 * Lc + l) * Hc * Hc;

      for (int ks = 0; ks < Hc / 4; ks += 32) {
        const int kb = k0 + ks;
        v16bf a  = load_a_frag_f32(scur, kb, Hc, lane);
        v16bf b0 = load_b_frag(W0, nbase, kb, Hc, lane);
        v16bf b1 = load_b_frag(W1, nbase, kb, Hc, lane);
        v16bf b2 = load_b_frag(W2, nbase, kb, Hc, lane);
        // keep next iteration's weights warm (global_prefetch_b8)
        __builtin_prefetch(W0 + (size_t)(nbase + (lane & 15)) * Hc + kb + 32, 0, 0);
        acc0 = wmma_bf16(a, b0, acc0);
        acc1 = wmma_bf16(a, b1, acc1);
        acc2 = wmma_bf16(a, b2, acc2);
      }

#pragma unroll
      for (int v = 0; v < 8; ++v) {
        red[wave][0][v * 32 + lane] = acc0[v];
        red[wave][1][v * 32 + lane] = acc1[v];
        red[wave][2][v * 32 + lane] = acc2[v];
      }
      __syncthreads();

      if (wave == 0) {
        const int n  = nbase + (lane & 15);
        const int hi = lane >> 4;
        const float bhv = bh[l * Hc + n];
        const float btv = bt[l * Hc + n];
        const float bcv = bc[l * Hc + n];
#pragma unroll
        for (int v = 0; v < 8; ++v) {
          const int m = v + hi * 8;
          float ph = bhv, pt = btv, pc = bcv;
#pragma unroll
          for (int w = 0; w < 4; ++w) {
            ph += red[w][0][v * 32 + lane];
            pt += red[w][1][v * 32 + lane];
            pc += red[w][2][v * 32 + lane];
          }
          if (l == 0) {
            const size_t pb = ((size_t)t * Bc + m) * Hc + n;
            ph += (float)proj[pb];
            pt += (float)proj[(size_t)Tc * Bc * Hc + pb];
            pc += (float)proj[2 * (size_t)Tc * Bc * Hc + pb];
          }
          const float sold = scur[m * Hc + n];
          const float hv = tanhf(ph);
          const float tv = 1.f / (1.f + __expf(-pt));
          const float cv = 1.f / (1.f + __expf(-pc));
          const float sn = hv * tv + sold * cv;
          snxt[m * Hc + n] = sn;
          if (l == Lc - 1) {
            out[((size_t)m * Tc + t) * Hc + n] = sn;  // [B,T,H]
            if (t == Tc - 1) sfin[m * Hc + n] = sn;
          }
        }
      }

      // ---- device-wide barrier between micro-layers ----
      __threadfence();   // make snxt stores device-visible
      __syncthreads();   // all lanes' fences done; also protects LDS reuse
      ++gen;
      if (threadIdx.x == 0) {
        __hip_atomic_fetch_add(counter, 1u, __ATOMIC_RELEASE,
                               __HIP_MEMORY_SCOPE_AGENT);
        const unsigned target = gen * NBLK;
        while (__hip_atomic_load(counter, __ATOMIC_ACQUIRE,
                                 __HIP_MEMORY_SCOPE_AGENT) < target)
          __builtin_amdgcn_s_sleep(1);
      }
      __syncthreads();
      __threadfence();   // acquire-side: drop stale cached state

      float* tmp = scur; scur = snxt; snxt = tmp;
    }
  }
}

// ---------------------------------------------------------------------------
// Launcher. Inputs (setup_inputs order):
//  0:x 1:s0 2:w_h 3:w_t 4:w_c 5:R_h 6:b_h 7:R_t 8:b_t 9:R_c 10:b_c
// Output: [B,T,H] f32 followed by [B,H] f32.
// Workspace (~139 MB): Rpack bf16 | wT bf16 | proj bf16 | s x2 f32 | counter.
// ---------------------------------------------------------------------------
extern "C" void kernel_launch(void* const* d_in, const int* in_sizes, int n_in,
                              void* d_out, int out_size, void* d_ws, size_t ws_size,
                              hipStream_t stream) {
  const float* x   = (const float*)d_in[0];
  const float* s0  = (const float*)d_in[1];
  const float* w_h = (const float*)d_in[2];
  const float* w_t = (const float*)d_in[3];
  const float* w_c = (const float*)d_in[4];
  const float* R_h = (const float*)d_in[5];
  const float* b_h = (const float*)d_in[6];
  const float* R_t = (const float*)d_in[7];
  const float* b_t = (const float*)d_in[8];
  const float* R_c = (const float*)d_in[9];
  const float* b_c = (const float*)d_in[10];

  float* out  = (float*)d_out;                 // [B,T,H]
  float* sfin = out + (size_t)Bc * Tc * Hc;    // [B,H]

  char* ws = (char*)d_ws;
  size_t off = 0;
  __bf16* Rpack = (__bf16*)(ws + off); off += (size_t)3 * Lc * Hc * Hc * 2;
  __bf16* wT    = (__bf16*)(ws + off); off += (size_t)3 * Hc * Ic * 2;
  __bf16* proj  = (__bf16*)(ws + off); off += (size_t)3 * Tc * Bc * Hc * 2;
  float*  sb0   = (float*)(ws + off);  off += (size_t)Bc * Hc * 4;
  float*  sb1   = (float*)(ws + off);  off += (size_t)Bc * Hc * 4;
  unsigned* counter = (unsigned*)(ws + off); off += 256;

  // Reset barrier counter + seed state (graph-capture safe async ops only).
  hipMemsetAsync(counter, 0, sizeof(unsigned), stream);
  hipMemcpyAsync(sb0, s0, (size_t)Bc * Hc * sizeof(float),
                 hipMemcpyDeviceToDevice, stream);

  pack_R<<<2048, 256, 0, stream>>>(R_h, R_t, R_c, Rpack);
  pack_wT<<<1024, 256, 0, stream>>>(w_h, w_t, w_c, wT);

  rhn_proj<<<dim3(Tc, 3, Hc / 64), 32, 0, stream>>>(x, wT, proj);

  rhn_recur<<<NBLK, RECUR_THREADS, 0, stream>>>(
      Rpack, proj, b_h, b_t, b_c, sb0, sb1, out, sfin, counter);
}
